// tree_message_passing2D_32117765439933
// MI455X (gfx1250) — matrix operand order repaired and checked by reading, not compiled, not run on hardware
//
#include <hip/hip_runtime.h>
#include <hip/hip_bf16.h>
#include <stdint.h>

// ---------------- problem constants (fixed by the reference) ----------------
#define B_TREES 64
#define DEPTH   11
#define NT      ((1 << (DEPTH + 1)) - 1)     // 4095 nodes per tree
#define NNODES  (B_TREES * NT)               // 262080
#define F_DIM   256
#define H_DIM   128

typedef __bf16 bf16_t;
typedef __attribute__((ext_vector_type(16))) __bf16 v16bf;
typedef __attribute__((ext_vector_type(8)))  float  v8f;

union ABFrag { float4 f4[2]; v16bf v; };

// f32 -> bf16 with round-to-nearest-even (bit trick; avoids relying on __bf16 casts)
__device__ __forceinline__ unsigned short f32_to_bf16_bits(float f) {
  unsigned int u = __float_as_uint(f);
  unsigned int r = u + 0x7FFFu + ((u >> 16) & 1u);
  return (unsigned short)(r >> 16);
}
__device__ __forceinline__ unsigned int pack_bf16x2(float a, float b) {
  return (unsigned int)f32_to_bf16_bits(a) | ((unsigned int)f32_to_bf16_bits(b) << 16);
}
__device__ __forceinline__ float sigmoidf_dev(float x) { return 1.0f / (1.0f + __expf(-x)); }

// ---- WMMA fragment loaders -------------------------------------------------
// A (16xK bf16, row-major tile in LDS, ldk elems/row), per ISA 16-bit A 16x32 layout:
// lane<16: VGPR0-3 K=0..7, VGPR4-7 K=16..23 ; lane>=16: K=8..15 and K=24..31, M = lane%16
__device__ __forceinline__ v16bf load_a_lds(const bf16_t* tile, int ldk, int m, int k0, int lane) {
  const bf16_t* row = tile + m * ldk + k0 + ((lane & 16) ? 8 : 0);
  ABFrag u;
  u.f4[0] = *(const float4*)(row);        // K run base .. base+7
  u.f4[1] = *(const float4*)(row + 16);   // K run base+16 .. base+23
  return u.v;
}
// B fragments pre-swizzled to fragment-major layout:
// packed[((ntile*ktiles + ktile)*32 + lane)*16 + e]; a wave's fragment is one
// contiguous 1KB block -> perfectly coalesced 2x b128 per lane.
__device__ __forceinline__ v16bf load_b_packed(const bf16_t* wp, int ntile, int ktiles,
                                               int ktile, int lane) {
  const bf16_t* p = wp + (((size_t)(ntile * ktiles + ktile) * 32) + lane) * 16;
  ABFrag u;
  u.f4[0] = *(const float4*)(p);
  u.f4[1] = *(const float4*)(p + 8);
  return u.v;
}
__device__ __forceinline__ v8f wmma_bf16(v16bf a, v16bf b, v8f c) {
  return __builtin_amdgcn_wmma_f32_16x16x32_bf16(false, a, false, b, (short)0, c, false, false);
}

// ---------------- weight f32 -> bf16 pack (fragment-major B layout) ---------
// B[k,n] = W[n,k]; WMMA 16-bit B 32x16: lanes 0-15: N=lane, K=k0..k0+7,k0+16..k0+23 ;
// lanes 16-31: N=lane-16, K=k0+8..k0+15,k0+24..k0+31 (2 K per VGPR).
__global__ void __launch_bounds__(256) pack_b_kernel(const float* __restrict__ w,
                                                     bf16_t* __restrict__ d,
                                                     int nout, int kdim) {
  const int g = blockIdx.x * 256 + threadIdx.x;      // (nt*KT + kt)*32 + lane
  const int KT = kdim >> 5;
  const int ngroups = (nout >> 4) * KT * 32;
  if (g >= ngroups) return;
  const int lane = g & 31;
  const int tile = g >> 5;
  const int nt = tile / KT, kt = tile - nt * KT;
  const int n = nt * 16 + (lane & 15);
  const int k = kt * 32 + ((lane & 16) ? 8 : 0);
  const float* src = w + (size_t)n * kdim + k;
  unsigned int* dst = (unsigned int*)(d + (size_t)g * 16);
#pragma unroll
  for (int e = 0; e < 8; ++e)
    dst[e] = pack_bf16x2(src[2 * e + ((e & 4) ? 8 : 0)], src[2 * e + 1 + ((e & 4) ? 8 : 0)]);
}

// ---------------- leaf init: h[leaf] = (1-z)*tanh(x@Wh.T), z = sig(x@Wz.T) ---
__global__ void __launch_bounds__(256) gru_leaf_kernel(const float* __restrict__ x,
    const bf16_t* __restrict__ wz, const bf16_t* __restrict__ wh,
    float* __restrict__ h)
{
  __shared__ alignas(16) bf16_t xs[16 * F_DIM];
  const int t = threadIdx.x, lane = t & 31, wv = t >> 5, blk = blockIdx.x;

  { // stage 16 leaf rows of x -> bf16 LDS
    const int r = t >> 4, c = (t & 15) * 16;
    const int li = blk * 16 + r;
    const int b = li >> DEPTH, j = li & ((1 << DEPTH) - 1);
    const size_t row = (size_t)b * NT + ((1 << DEPTH) - 1) + j;
    const float4* xp = (const float4*)(x + row * F_DIM + c);
    float4 f0 = xp[0], f1 = xp[1], f2 = xp[2], f3 = xp[3];
    unsigned int* d = (unsigned int*)&xs[r * F_DIM + c];
    d[0] = pack_bf16x2(f0.x, f0.y); d[1] = pack_bf16x2(f0.z, f0.w);
    d[2] = pack_bf16x2(f1.x, f1.y); d[3] = pack_bf16x2(f1.z, f1.w);
    d[4] = pack_bf16x2(f2.x, f2.y); d[5] = pack_bf16x2(f2.z, f2.w);
    d[6] = pack_bf16x2(f3.x, f3.y); d[7] = pack_bf16x2(f3.z, f3.w);
  }
  __syncthreads();

  const int nn = wv * 16 + (lane & 15), m = lane & 15;
  v8f accZ = {}, accH = {};
#pragma unroll
  for (int kt = 0; kt < 8; ++kt) {
    v16bf a = load_a_lds(xs, F_DIM, m, kt * 32, lane);
    accZ = wmma_bf16(a, load_b_packed(wz, wv, 8, kt, lane), accZ);
    accH = wmma_bf16(a, load_b_packed(wh, wv, 8, kt, lane), accH);
  }
#pragma unroll
  for (int jj = 0; jj < 8; ++jj) {
    const int mm = (lane < 16) ? jj : jj + 8;
    const int li = blk * 16 + mm;
    const int b = li >> DEPTH, j = li & ((1 << DEPTH) - 1);
    const size_t row = (size_t)b * NT + ((1 << DEPTH) - 1) + j;
    float z = sigmoidf_dev(accZ[jj]);
    h[row * H_DIM + nn] = (1.0f - z) * tanhf(accH[jj]);
  }
}

// ---------------- bottom-up level step (fathers at level k) -----------------
__global__ void __launch_bounds__(256) gru_up_kernel(const float* __restrict__ x,
    const bf16_t* __restrict__ wz, const bf16_t* __restrict__ wr, const bf16_t* __restrict__ wh,
    const bf16_t* __restrict__ uz, const bf16_t* __restrict__ ur, const bf16_t* __restrict__ uh,
    const float* __restrict__ bz, const float* __restrict__ br,
    float* __restrict__ h, int k)
{
  __shared__ alignas(16) bf16_t xs[16 * F_DIM];
  __shared__ alignas(16) float  sf[16 * H_DIM];
  __shared__ alignas(16) bf16_t sb[16 * H_DIM];
  __shared__ alignas(16) bf16_t rs[16 * H_DIM];
  const int t = threadIdx.x, lane = t & 31, wv = t >> 5, blk = blockIdx.x;
  const int off = (1 << k) - 1, offc = (1 << (k + 1)) - 1, mask = (1 << k) - 1;

  { // stage xf (16x256) and s = h[child0]+h[child1] (16x128, f32 + bf16)
    const int r = t >> 4;
    const int fi = blk * 16 + r;
    const int b = fi >> k, j = fi & mask;
    const size_t frow = (size_t)b * NT + off + j;
    const int c = (t & 15) * 16;
    const float4* xp = (const float4*)(x + frow * F_DIM + c);
    float4 f0 = xp[0], f1 = xp[1], f2 = xp[2], f3 = xp[3];
    unsigned int* d = (unsigned int*)&xs[r * F_DIM + c];
    d[0] = pack_bf16x2(f0.x, f0.y); d[1] = pack_bf16x2(f0.z, f0.w);
    d[2] = pack_bf16x2(f1.x, f1.y); d[3] = pack_bf16x2(f1.z, f1.w);
    d[4] = pack_bf16x2(f2.x, f2.y); d[5] = pack_bf16x2(f2.z, f2.w);
    d[6] = pack_bf16x2(f3.x, f3.y); d[7] = pack_bf16x2(f3.z, f3.w);

    const int c8 = (t & 15) * 8;
    const size_t srow0 = (size_t)b * NT + offc + 2 * j;
    const float4* h0 = (const float4*)(h + srow0 * H_DIM + c8);
    const float4* h1 = (const float4*)(h + (srow0 + 1) * H_DIM + c8);
    float4 p0 = h0[0], p1 = h0[1], q0 = h1[0], q1 = h1[1];
    float4 s0, s1;
    s0.x = p0.x + q0.x; s0.y = p0.y + q0.y; s0.z = p0.z + q0.z; s0.w = p0.w + q0.w;
    s1.x = p1.x + q1.x; s1.y = p1.y + q1.y; s1.z = p1.z + q1.z; s1.w = p1.w + q1.w;
    float4* sfp = (float4*)&sf[r * H_DIM + c8];
    sfp[0] = s0; sfp[1] = s1;
    unsigned int* sd = (unsigned int*)&sb[r * H_DIM + c8];
    sd[0] = pack_bf16x2(s0.x, s0.y); sd[1] = pack_bf16x2(s0.z, s0.w);
    sd[2] = pack_bf16x2(s1.x, s1.y); sd[3] = pack_bf16x2(s1.z, s1.w);
  }
  __syncthreads();

  const int nn = wv * 16 + (lane & 15), m = lane & 15;
  v8f accZ = {}, accR = {};
#pragma unroll
  for (int kt = 0; kt < 8; ++kt) {
    v16bf a = load_a_lds(xs, F_DIM, m, kt * 32, lane);
    accZ = wmma_bf16(a, load_b_packed(wz, wv, 8, kt, lane), accZ);
    accR = wmma_bf16(a, load_b_packed(wr, wv, 8, kt, lane), accR);
  }
#pragma unroll
  for (int kt = 0; kt < 4; ++kt) {
    v16bf a = load_a_lds(sb, H_DIM, m, kt * 32, lane);
    accZ = wmma_bf16(a, load_b_packed(uz, wv, 4, kt, lane), accZ);
    accR = wmma_bf16(a, load_b_packed(ur, wv, 4, kt, lane), accR);
  }
  const float bzv = bz[nn], brv = br[nn];
  float zv[8], sv[8];
#pragma unroll
  for (int jj = 0; jj < 8; ++jj) {
    const int mm = (lane < 16) ? jj : jj + 8;
    sv[jj] = sf[mm * H_DIM + nn];
    zv[jj] = sigmoidf_dev(accZ[jj] + bzv);
    float rv = sigmoidf_dev(accR[jj] + brv);
    ((unsigned short*)rs)[mm * H_DIM + nn] = f32_to_bf16_bits(rv * sv[jj]);
  }
  __syncthreads();

  v8f accH = {};
#pragma unroll
  for (int kt = 0; kt < 8; ++kt)
    accH = wmma_bf16(load_a_lds(xs, F_DIM, m, kt * 32, lane),
                     load_b_packed(wh, wv, 8, kt, lane), accH);
#pragma unroll
  for (int kt = 0; kt < 4; ++kt)
    accH = wmma_bf16(load_a_lds(rs, H_DIM, m, kt * 32, lane),
                     load_b_packed(uh, wv, 4, kt, lane), accH);
#pragma unroll
  for (int jj = 0; jj < 8; ++jj) {
    const int mm = (lane < 16) ? jj : jj + 8;
    const int fi = blk * 16 + mm;
    const int b = fi >> k, j = fi & mask;
    const size_t frow = (size_t)b * NT + off + j;
    float hh = tanhf(accH[jj]);
    h[frow * H_DIM + nn] = zv[jj] * sv[jj] + (1.0f - zv[jj]) * hh;
  }
}

// ---------------- root copy: hd[root] = h[root] -----------------------------
__global__ void __launch_bounds__(256) root_copy_kernel(const float* __restrict__ h,
                                                        float* __restrict__ hd) {
  int i = blockIdx.x * 256 + threadIdx.x;
  if (i < B_TREES * H_DIM) {
    int b = i >> 7, c = i & 127;
    size_t row = (size_t)b * NT;
    hd[row * H_DIM + c] = h[row * H_DIM + c];
  }
}

// ---------------- top-down level step (sons at level k+1) -------------------
__global__ void __launch_bounds__(256) gru_down_kernel(const float* __restrict__ x,
    const bf16_t* __restrict__ wz, const bf16_t* __restrict__ wr, const bf16_t* __restrict__ wh,
    const bf16_t* __restrict__ uz, const bf16_t* __restrict__ ur, const bf16_t* __restrict__ uh,
    const float* __restrict__ bz, const float* __restrict__ br,
    float* __restrict__ hd, int k)
{
  __shared__ alignas(16) bf16_t xs[16 * F_DIM];
  __shared__ alignas(16) float  sf[16 * H_DIM];
  __shared__ alignas(16) bf16_t sb[16 * H_DIM];
  __shared__ alignas(16) bf16_t rs[16 * H_DIM];
  const int t = threadIdx.x, lane = t & 31, wv = t >> 5, blk = blockIdx.x;
  const int off = (1 << k) - 1, offc = (1 << (k + 1)) - 1, maskc = (1 << (k + 1)) - 1;

  { // stage xs (son rows of x) and s = hd[father-of-son]
    const int r = t >> 4;
    const int si = blk * 16 + r;
    const int b = si >> (k + 1), j = si & maskc;
    const size_t srow = (size_t)b * NT + offc + j;
    const size_t frow = (size_t)b * NT + off + (j >> 1);
    const int c = (t & 15) * 16;
    const float4* xp = (const float4*)(x + srow * F_DIM + c);
    float4 f0 = xp[0], f1 = xp[1], f2 = xp[2], f3 = xp[3];
    unsigned int* d = (unsigned int*)&xs[r * F_DIM + c];
    d[0] = pack_bf16x2(f0.x, f0.y); d[1] = pack_bf16x2(f0.z, f0.w);
    d[2] = pack_bf16x2(f1.x, f1.y); d[3] = pack_bf16x2(f1.z, f1.w);
    d[4] = pack_bf16x2(f2.x, f2.y); d[5] = pack_bf16x2(f2.z, f2.w);
    d[6] = pack_bf16x2(f3.x, f3.y); d[7] = pack_bf16x2(f3.z, f3.w);

    const int c8 = (t & 15) * 8;
    const float4* hp = (const float4*)(hd + frow * H_DIM + c8);
    float4 s0 = hp[0], s1 = hp[1];
    float4* sfp = (float4*)&sf[r * H_DIM + c8];
    sfp[0] = s0; sfp[1] = s1;
    unsigned int* sd = (unsigned int*)&sb[r * H_DIM + c8];
    sd[0] = pack_bf16x2(s0.x, s0.y); sd[1] = pack_bf16x2(s0.z, s0.w);
    sd[2] = pack_bf16x2(s1.x, s1.y); sd[3] = pack_bf16x2(s1.z, s1.w);
  }
  __syncthreads();

  const int nn = wv * 16 + (lane & 15), m = lane & 15;
  v8f accZ = {}, accR = {};
#pragma unroll
  for (int kt = 0; kt < 8; ++kt) {
    v16bf a = load_a_lds(xs, F_DIM, m, kt * 32, lane);
    accZ = wmma_bf16(a, load_b_packed(wz, wv, 8, kt, lane), accZ);
    accR = wmma_bf16(a, load_b_packed(wr, wv, 8, kt, lane), accR);
  }
#pragma unroll
  for (int kt = 0; kt < 4; ++kt) {
    v16bf a = load_a_lds(sb, H_DIM, m, kt * 32, lane);
    accZ = wmma_bf16(a, load_b_packed(uz, wv, 4, kt, lane), accZ);
    accR = wmma_bf16(a, load_b_packed(ur, wv, 4, kt, lane), accR);
  }
  const float bzv = bz[nn], brv = br[nn];
  float zv[8], sv[8];
#pragma unroll
  for (int jj = 0; jj < 8; ++jj) {
    const int mm = (lane < 16) ? jj : jj + 8;
    sv[jj] = sf[mm * H_DIM + nn];
    zv[jj] = sigmoidf_dev(accZ[jj] + bzv);
    float rv = sigmoidf_dev(accR[jj] + brv);
    ((unsigned short*)rs)[mm * H_DIM + nn] = f32_to_bf16_bits(rv * sv[jj]);
  }
  __syncthreads();

  v8f accH = {};
#pragma unroll
  for (int kt = 0; kt < 8; ++kt)
    accH = wmma_bf16(load_a_lds(xs, F_DIM, m, kt * 32, lane),
                     load_b_packed(wh, wv, 8, kt, lane), accH);
#pragma unroll
  for (int kt = 0; kt < 4; ++kt)
    accH = wmma_bf16(load_a_lds(rs, H_DIM, m, kt * 32, lane),
                     load_b_packed(uh, wv, 4, kt, lane), accH);
#pragma unroll
  for (int jj = 0; jj < 8; ++jj) {
    const int mm = (lane < 16) ? jj : jj + 8;
    const int si = blk * 16 + mm;
    const int b = si >> (k + 1), j = si & maskc;
    const size_t srow = (size_t)b * NT + offc + j;
    float hh = tanhf(accH[jj]);
    hd[srow * H_DIM + nn] = zv[jj] * sv[jj] + (1.0f - zv[jj]) * hh;
  }
}

// ---------------- out = h@F1w.T + F1b + hd@F2w.T + F2b ----------------------
__global__ void __launch_bounds__(256) out_kernel(const float* __restrict__ h,
    const float* __restrict__ hd,
    const bf16_t* __restrict__ f1w, const bf16_t* __restrict__ f2w,
    const float* __restrict__ f1b, const float* __restrict__ f2b,
    float* __restrict__ out)
{
  __shared__ alignas(16) bf16_t hs[16 * H_DIM];
  __shared__ alignas(16) bf16_t hds[16 * H_DIM];
  const int t = threadIdx.x, lane = t & 31, wv = t >> 5;
  const size_t m0 = (size_t)blockIdx.x * 16;

  {
    const int r = t >> 4, c8 = (t & 15) * 8;
    const float4* hp  = (const float4*)(h  + (m0 + r) * H_DIM + c8);
    const float4* hdp = (const float4*)(hd + (m0 + r) * H_DIM + c8);
    float4 a0 = hp[0],  a1 = hp[1];
    float4 b0 = hdp[0], b1 = hdp[1];
    unsigned int* d0 = (unsigned int*)&hs[r * H_DIM + c8];
    d0[0] = pack_bf16x2(a0.x, a0.y); d0[1] = pack_bf16x2(a0.z, a0.w);
    d0[2] = pack_bf16x2(a1.x, a1.y); d0[3] = pack_bf16x2(a1.z, a1.w);
    unsigned int* d1 = (unsigned int*)&hds[r * H_DIM + c8];
    d1[0] = pack_bf16x2(b0.x, b0.y); d1[1] = pack_bf16x2(b0.z, b0.w);
    d1[2] = pack_bf16x2(b1.x, b1.y); d1[3] = pack_bf16x2(b1.z, b1.w);
  }
  __syncthreads();

  const int m = lane & 15;
#pragma unroll
  for (int tt = 0; tt < 2; ++tt) {
    const int nt = wv * 2 + tt;
    const int nn = nt * 16 + (lane & 15);
    v8f acc = {};
#pragma unroll
    for (int kt = 0; kt < 4; ++kt)
      acc = wmma_bf16(load_a_lds(hs, H_DIM, m, kt * 32, lane),
                      load_b_packed(f1w, nt, 4, kt, lane), acc);
#pragma unroll
    for (int kt = 0; kt < 4; ++kt)
      acc = wmma_bf16(load_a_lds(hds, H_DIM, m, kt * 32, lane),
                      load_b_packed(f2w, nt, 4, kt, lane), acc);
    const float bias = f1b[nn] + f2b[nn];
#pragma unroll
    for (int jj = 0; jj < 8; ++jj) {
      const int mm = (lane < 16) ? jj : jj + 8;
      out[(m0 + mm) * F_DIM + nn] = acc[jj] + bias;
    }
  }
}

// ---------------- host driver ----------------------------------------------
extern "C" void kernel_launch(void* const* d_in, const int* in_sizes, int n_in,
                              void* d_out, int out_size, void* d_ws, size_t ws_size,
                              hipStream_t stream)
{
  const float* x   = (const float*)d_in[0];
  const float* Wh  = (const float*)d_in[1];
  const float* Wz  = (const float*)d_in[2];
  const float* Wr  = (const float*)d_in[3];
  const float* Uh  = (const float*)d_in[4];
  const float* Uz  = (const float*)d_in[5];
  const float* Ur  = (const float*)d_in[6];
  const float* bz  = (const float*)d_in[7];
  const float* br  = (const float*)d_in[8];
  const float* Whd = (const float*)d_in[9];
  const float* Wzd = (const float*)d_in[10];
  const float* Wrd = (const float*)d_in[11];
  const float* Uhd = (const float*)d_in[12];
  const float* Uzd = (const float*)d_in[13];
  const float* Urd = (const float*)d_in[14];
  const float* bzd = (const float*)d_in[15];
  const float* brd = (const float*)d_in[16];
  const float* F1w = (const float*)d_in[17];
  const float* F1b = (const float*)d_in[18];
  const float* F2w = (const float*)d_in[19];
  const float* F2b = (const float*)d_in[20];
  float* out = (float*)d_out;

  // workspace layout: h | hd | packed bf16 weights
  float* h  = (float*)d_ws;
  float* hd = h + (size_t)NNODES * H_DIM;
  bf16_t* wb = (bf16_t*)(hd + (size_t)NNODES * H_DIM);
  const int WF = H_DIM * F_DIM;   // 32768 (8 n-tiles x 8 k-tiles)
  const int UU = H_DIM * H_DIM;   // 16384 (8 n-tiles x 4 k-tiles)
  const int FH = F_DIM * H_DIM;   // 32768 (16 n-tiles x 4 k-tiles)
  bf16_t* bWz  = wb;         bf16_t* bWr  = bWz  + WF;  bf16_t* bWh  = bWr  + WF;
  bf16_t* bUz  = bWh  + WF;  bf16_t* bUr  = bUz  + UU;  bf16_t* bUh  = bUr  + UU;
  bf16_t* bWzd = bUh  + UU;  bf16_t* bWrd = bWzd + WF;  bf16_t* bWhd = bWrd + WF;
  bf16_t* bUzd = bWhd + WF;  bf16_t* bUrd = bUzd + UU;  bf16_t* bUhd = bUrd + UU;
  bf16_t* bF1  = bUhd + UU;  bf16_t* bF2  = bF1  + FH;

  auto pack = [&](const float* s, bf16_t* d, int nout, int kdim) {
    const int ngroups = (nout / 16) * (kdim / 32) * 32;
    pack_b_kernel<<<(ngroups + 255) / 256, 256, 0, stream>>>(s, d, nout, kdim);
  };
  pack(Wz,  bWz,  H_DIM, F_DIM); pack(Wr,  bWr,  H_DIM, F_DIM); pack(Wh,  bWh,  H_DIM, F_DIM);
  pack(Uz,  bUz,  H_DIM, H_DIM); pack(Ur,  bUr,  H_DIM, H_DIM); pack(Uh,  bUh,  H_DIM, H_DIM);
  pack(Wzd, bWzd, H_DIM, F_DIM); pack(Wrd, bWrd, H_DIM, F_DIM); pack(Whd, bWhd, H_DIM, F_DIM);
  pack(Uzd, bUzd, H_DIM, H_DIM); pack(Urd, bUrd, H_DIM, H_DIM); pack(Uhd, bUhd, H_DIM, H_DIM);
  pack(F1w, bF1,  F_DIM, H_DIM); pack(F2w, bF2,  F_DIM, H_DIM);

  // bottom-up
  gru_leaf_kernel<<<(B_TREES << DEPTH) / 16, 256, 0, stream>>>(x, bWz, bWh, h);
  for (int k = DEPTH - 1; k >= 0; --k)
    gru_up_kernel<<<(B_TREES << k) / 16, 256, 0, stream>>>(
        x, bWz, bWr, bWh, bUz, bUr, bUh, bz, br, h, k);

  // top-down
  root_copy_kernel<<<(B_TREES * H_DIM + 255) / 256, 256, 0, stream>>>(h, hd);
  for (int k = 0; k < DEPTH; ++k)
    gru_down_kernel<<<(B_TREES << (k + 1)) / 16, 256, 0, stream>>>(
        x, bWzd, bWrd, bWhd, bUzd, bUrd, bUhd, bzd, brd, hd, k);

  // output projection
  out_kernel<<<NNODES / 16, 256, 0, stream>>>(h, hd, bF1, bF2, F1b, F2b, out);
}
